// W8A8B32O8LinearForMac_61297773248988
// MI455X (gfx1250) — 8-bit matrix operand order repaired and checked by reading, compiled, not run on hardware
//
#include <hip/hip_runtime.h>

#define TOKENS 8192
#define IN_F   4096
#define OUT_F  4096

typedef __attribute__((ext_vector_type(2))) int v2i;
typedef __attribute__((ext_vector_type(4))) int v4i;
typedef __attribute__((ext_vector_type(8))) int v8i;

__global__ __launch_bounds__(256) void pack_x_kernel(const int* __restrict__ xi,
                                                     int* __restrict__ xq,
                                                     int n_dwords) {
    int i = blockIdx.x * blockDim.x + threadIdx.x;
    if (i >= n_dwords) return;
    const int* p = xi + (long)i * 4;
    int d = (p[0] & 0xFF) | ((p[1] & 0xFF) << 8) | ((p[2] & 0xFF) << 16)
          | (p[3] << 24);
    xq[i] = d;
}

__global__ __launch_bounds__(256) void pack_w_kernel(const int* __restrict__ wi,
                                                     int* __restrict__ wq) {
    long idx = (long)blockIdx.x * blockDim.x + threadIdx.x;
    int n  = (int)(idx & (OUT_F - 1));
    int k4 = (int)(idx >> 12);
    if (k4 >= IN_F / 4) return;
    long k = (long)k4 * 4;
    int b0 = wi[(k + 0) * OUT_F + n] & 0xFF;
    int b1 = wi[(k + 1) * OUT_F + n] & 0xFF;
    int b2 = wi[(k + 2) * OUT_F + n] & 0xFF;
    int b3 = wi[(k + 3) * OUT_F + n];
    wq[(long)n * (IN_F / 4) + k4] = b0 | (b1 << 8) | (b2 << 16) | (b3 << 24);
}

__global__ __launch_bounds__(256) void w8a8_gemm_kernel(
        const signed char* __restrict__ xq,
        const signed char* __restrict__ wq,
        const int*  __restrict__ bias,
        const float* __restrict__ scales,
        float* __restrict__ out)
{
    const int lane  = threadIdx.x & 31;
    const int wave  = threadIdx.x >> 5;
    const int mw    = wave >> 2;
    const int nw    = wave & 3;
    const int mWave = blockIdx.y * 64  + mw * 32;
    const int nWave = blockIdx.x * 256 + nw * 64;

    const int half  = lane >> 4;
    const int l15   = lane & 15;

    const signed char* aRow0 = xq + (long)(mWave + l15) * IN_F + half * 16;
    const signed char* aRow1 = aRow0 + 16L * IN_F;

    const signed char* bBase = wq + (long)(nWave + l15) * IN_F + half * 16;

    v8i acc[8];
    #pragma unroll
    for (int t = 0; t < 8; ++t) acc[t] = (v8i){0, 0, 0, 0, 0, 0, 0, 0};

    for (int k0 = 0; k0 < IN_F; k0 += 64) {
        const signed char* ap0 = aRow0 + k0;
        const signed char* ap1 = aRow1 + k0;
        __builtin_prefetch(ap0 + 64, 0, 3);
        __builtin_prefetch(ap1 + 64, 0, 3);
        v2i a00 = *(const v2i*)(ap0 + 0);
        v2i a01 = *(const v2i*)(ap0 + 8);
        v2i a02 = *(const v2i*)(ap0 + 32);
        v2i a03 = *(const v2i*)(ap0 + 40);
        v2i a10 = *(const v2i*)(ap1 + 0);
        v2i a11 = *(const v2i*)(ap1 + 8);
        v2i a12 = *(const v2i*)(ap1 + 32);
        v2i a13 = *(const v2i*)(ap1 + 40);
        v8i A0, A1;
        A0[0] = a00[0]; A0[1] = a00[1]; A0[2] = a01[0]; A0[3] = a01[1];
        A0[4] = a02[0]; A0[5] = a02[1]; A0[6] = a03[0]; A0[7] = a03[1];
        A1[0] = a10[0]; A1[1] = a10[1]; A1[2] = a11[0]; A1[3] = a11[1];
        A1[4] = a12[0]; A1[5] = a12[1]; A1[6] = a13[0]; A1[7] = a13[1];

        #pragma unroll
        for (int t = 0; t < 4; ++t) {
            const signed char* bp = bBase + (long)t * 16 * IN_F + k0;
            v4i b0 = *(const v4i*)(bp + 0);
            v4i b1 = *(const v4i*)(bp + 32);
            v8i B;
            B[0] = b0[0]; B[1] = b0[1]; B[2] = b0[2]; B[3] = b0[3];
            B[4] = b1[0]; B[5] = b1[1]; B[6] = b1[2]; B[7] = b1[3];
            acc[t] = __builtin_amdgcn_wmma_i32_16x16x64_iu8(
                         true, A0, true, B, acc[t], false, false);
            acc[t + 4] = __builtin_amdgcn_wmma_i32_16x16x64_iu8(
                         true, A1, true, B, acc[t + 4], false, false);
        }
    }

    #pragma unroll
    for (int t = 0; t < 4; ++t) {
        const int col = nWave + t * 16 + l15;
        const float s = scales[col] * 20.0f;
        const float b = (float)bias[col];
        #pragma unroll
        for (int u = 0; u < 2; ++u) {
            const int rowBase = mWave + u * 16 + half * 8;
            const v8i a = acc[t + 4 * u];
            #pragma unroll
            for (int r = 0; r < 8; ++r) {
                float v = ((float)a[r] + b) * s;
                v = fminf(fmaxf(v, -128.0f), 127.0f);
                int q = (int)v;
                out[(long)(rowBase + r) * OUT_F + col] = (float)q;
            }
        }
    }
}

extern "C" void kernel_launch(void* const* d_in, const int* in_sizes, int n_in,
                              void* d_out, int out_size, void* d_ws, size_t ws_size,
                              hipStream_t stream) {
    const int*   xi     = (const int*)  d_in[0];
    const int*   wi     = (const int*)  d_in[1];
    const int*   bias   = (const int*)  d_in[2];
    const float* scales = (const float*)d_in[3];
    float*       out    = (float*)d_out;

    signed char* xq = (signed char*)d_ws;
    signed char* wq = xq + (size_t)TOKENS * IN_F;

    const int xDwords = TOKENS * IN_F / 4;
    pack_x_kernel<<<(xDwords + 255) / 256, 256, 0, stream>>>(xi, (int*)xq, xDwords);

    const long wDwords = (long)OUT_F * (IN_F / 4);
    pack_w_kernel<<<(unsigned)((wDwords + 255) / 256), 256, 0, stream>>>(wi, (int*)wq);

    dim3 grid(OUT_F / 256, TOKENS / 64);
    w8a8_gemm_kernel<<<grid, 256, 0, stream>>>(xq, wq, bias, scales, out);
}
